// GemmaAttention_489626271771
// MI455X (gfx1250) — compile-verified, matrix operand order checked by reading, not run on hardware
//
#include <hip/hip_runtime.h>
#include <hip/hip_bf16.h>
#include <cstdint>
#include <cstddef>

// ---------------- problem constants ----------------
#define S_LEN    2048
#define HIDDEN_  3584
#define NH_      16
#define NKV_     8
#define HD_      256
#define QKV_OUT  8192          // (NH + 2*NKV) * HD
#define O_IN     4096          // NH * HD
#define SCALING_ 0.0625f       // 256^-0.5
#define SOFTCAP_ 50.0f
#define WINDOW_  1024
#define EPS_     1e-6f
#define MASKV    (-2.3819763e38f)

typedef __attribute__((ext_vector_type(16))) __bf16 v16bf;
typedef __attribute__((ext_vector_type(8)))  __bf16 v8bf;
typedef __attribute__((ext_vector_type(8)))  float  v8f;

// ---------------- helpers ----------------
__device__ inline float red_sum16(float v) {
  #pragma unroll
  for (int d = 1; d < 16; d <<= 1) v += __shfl_xor(v, d, 32);
  return v;
}
__device__ inline float red_max16(float v) {
  #pragma unroll
  for (int d = 1; d < 16; d <<= 1) v = fmaxf(v, __shfl_xor(v, d, 32));
  return v;
}
__device__ inline float red_sum32(float v) {
  #pragma unroll
  for (int d = 1; d < 32; d <<= 1) v += __shfl_xor(v, d, 32);
  return v;
}

// Build a 16-element bf16 A/B fragment from two contiguous 8-element halves.
// ISA 16-bit 16x32 fragment: lanes 0-15 hold K 0..7 (vgpr0-3) and 16..23
// (vgpr4-7); lanes 16-31 hold K 8..15 and 24..31.
__device__ inline v16bf load_frag(const __bf16* p0, const __bf16* p1) {
  v8bf a = *(const v8bf*)p0;
  v8bf b = *(const v8bf*)p1;
  v16bf r;
  #pragma unroll
  for (int i = 0; i < 8; ++i) { r[i] = a[i]; r[8 + i] = b[i]; }
  return r;
}

// Async DMA of one 16B chunk per lane: global (saddr + 32-bit voffset) -> LDS.
__device__ inline void async_b128(uint32_t lds_off, uint32_t g_off,
                                  const void* sbase) {
  asm volatile("global_load_async_to_lds_b128 %0, %1, %2"
               :: "v"(lds_off), "v"(g_off), "s"(sbase) : "memory");
}
__device__ inline void wait_async4() {
  asm volatile("s_wait_asynccnt 4" ::: "memory");
}
__device__ inline void wait_async0() {
  asm volatile("s_wait_asynccnt 0" ::: "memory");
}

// ---------------- fp32 -> bf16 convert ----------------
__global__ __launch_bounds__(256)
void cvt_bf16_kernel(const float* __restrict__ in, __bf16* __restrict__ out, int n) {
  for (int i = blockIdx.x * blockDim.x + threadIdx.x; i < n; i += gridDim.x * blockDim.x)
    out[i] = (__bf16)in[i];
}

// ---------------- bf16 WMMA GEMM: C[M,N] = A[M,K] * W[N,K]^T ----------------
// 256 threads = 8 waves in a 4(M) x 2(N) grid. Block tile 128x128, K-step 32.
// A/B tiles staged in LDS by global_load_async_to_lds_b128, double buffered.
// Each wave computes a 32x64 sub-tile (2x4 accumulators).
#define TM 128
#define TN 128
#define TK 32

__global__ __launch_bounds__(256)
void gemm_bf16_kernel(const __bf16* __restrict__ A, const __bf16* __restrict__ W,
                      float* __restrict__ C, int M, int N, int K) {
  __shared__ __bf16 sA[2][TM * TK];   // [m][k] row-major, 8KB per buffer
  __shared__ __bf16 sB[2][TN * TK];   // [n][k] row-major, 8KB per buffer

  const int tid  = threadIdx.x;
  const int lane = tid & 31;
  const int wid  = tid >> 5;
  const int wm   = wid >> 1;           // 0..3 (M direction)
  const int wn   = wid & 1;            // 0..1 (N direction)
  const int m0   = blockIdx.y * TM;
  const int n0   = blockIdx.x * TN;
  const int rlo  = lane & 15;
  const bool hi  = lane >= 16;
  const int kA0  = hi ? 8 : 0;
  const int kA1  = hi ? 24 : 16;

  const __bf16* baseA = A + (size_t)m0 * K;   // uniform
  const __bf16* baseB = W + (size_t)n0 * K;   // uniform

  // flat-LDS aperture: low 32 bits of generic shared pointer = LDS byte offset
  const uint32_t ldsA[2] = { (uint32_t)(uintptr_t)&sA[0][0],
                             (uint32_t)(uintptr_t)&sA[1][0] };
  const uint32_t ldsB[2] = { (uint32_t)(uintptr_t)&sB[0][0],
                             (uint32_t)(uintptr_t)&sB[1][0] };

  // Stage one K-step: 512 x 16B chunks for A, 512 for B; 4 chunks/thread.
  // chunk c: row = c>>2, col8 = (c&3)*8 bf16 elems.
  auto issue = [&](int buf, int k) {
    #pragma unroll
    for (int j = 0; j < 2; ++j) {
      const int c   = tid + 256 * j;
      const int row = c >> 2;
      const int col = (c & 3) * 8;
      const uint32_t goff = (uint32_t)(((uint32_t)row * (uint32_t)K + k + col) * 2u);
      async_b128(ldsA[buf] + (uint32_t)c * 16u, goff, baseA);
      async_b128(ldsB[buf] + (uint32_t)c * 16u, goff, baseB);
    }
  };

  v8f acc[2][4];
  #pragma unroll
  for (int t = 0; t < 2; ++t)
    #pragma unroll
    for (int nt = 0; nt < 4; ++nt) acc[t][nt] = (v8f){};

  const int nk = K / TK;
  issue(0, 0);

  for (int i = 0; i < nk; ++i) {
    const int buf = i & 1;
    const bool more = (i + 1) < nk;
    if (more) issue(buf ^ 1, (i + 1) * TK);      // overlap DMA with compute
    if (more) wait_async4();                     // only current buffer landed
    else      wait_async0();
    __syncthreads();

    const __bf16* a0 = &sA[buf][0];
    const __bf16* b0 = &sB[buf][0];

    v16bf af[2], bfr[4];
    #pragma unroll
    for (int t = 0; t < 2; ++t) {
      const int row = wm * 32 + t * 16 + rlo;
      af[t] = load_frag(a0 + row * TK + kA0, a0 + row * TK + kA1);
    }
    #pragma unroll
    for (int nt = 0; nt < 4; ++nt) {
      const int col = wn * 64 + nt * 16 + rlo;
      bfr[nt] = load_frag(b0 + col * TK + kA0, b0 + col * TK + kA1);
    }
    #pragma unroll
    for (int t = 0; t < 2; ++t)
      #pragma unroll
      for (int nt = 0; nt < 4; ++nt)
        acc[t][nt] = __builtin_amdgcn_wmma_f32_16x16x32_bf16(
            false, af[t], false, bfr[nt], (short)0, acc[t][nt], false, false);

    __syncthreads();   // everyone done reading buf before it is re-issued
  }

  // C/D layout: element r of lane l = C[(l<16 ? r : r+8)][l&15]
  #pragma unroll
  for (int t = 0; t < 2; ++t)
    #pragma unroll
    for (int nt = 0; nt < 4; ++nt)
      #pragma unroll
      for (int r = 0; r < 8; ++r) {
        const int row = m0 + wm * 32 + t * 16 + (hi ? r + 8 : r);
        const int col = n0 + wn * 64 + nt * 16 + rlo;
        C[(size_t)row * N + col] = acc[t][nt][r];
      }
}

// ---------------- RMS-norm + RoPE + layout prep ----------------
__global__ __launch_bounds__(256)
void prep_kernel(const float* __restrict__ qkv,
                 const float* __restrict__ qnw, const float* __restrict__ knw,
                 const float* __restrict__ fcos, const float* __restrict__ fsin,
                 __bf16* __restrict__ qb, __bf16* __restrict__ kb,
                 __bf16* __restrict__ vt) {
  const int s    = blockIdx.x;
  const int lane = threadIdx.x & 31;
  const int wid  = threadIdx.x >> 5;
  const float* row = qkv + (size_t)s * QKV_OUT;
  const float* fc  = fcos + (size_t)s * (HD_ / 2);
  const float* fs  = fsin + (size_t)s * (HD_ / 2);

  #pragma unroll
  for (int rep = 0; rep < 2; ++rep) {
    const int hq = wid + rep * 8;
    const float* hrow = row + hq * HD_;
    float x1[4], x2[4];
    float ss = 0.f;
    #pragma unroll
    for (int t = 0; t < 4; ++t) {
      int d = lane * 4 + t;
      x1[t] = hrow[d];
      x2[t] = hrow[d + 128];
      ss += x1[t] * x1[t] + x2[t] * x2[t];
    }
    ss = red_sum32(ss);
    float rstd = rsqrtf(ss * (1.0f / HD_) + EPS_);
    __bf16* dst = qb + ((size_t)hq * S_LEN + s) * HD_;
    #pragma unroll
    for (int t = 0; t < 4; ++t) {
      int d = lane * 4 + t;
      float a = x1[t] * rstd * (1.0f + qnw[d]);
      float b = x2[t] * rstd * (1.0f + qnw[d + 128]);
      float c = fc[d], sn = fs[d];
      dst[d]       = (__bf16)((a * c - b * sn) * SCALING_);
      dst[d + 128] = (__bf16)((a * sn + b * c) * SCALING_);
    }
  }

  {
    const int hk = wid;
    const float* hrow = row + NH_ * HD_ + hk * HD_;
    float x1[4], x2[4];
    float ss = 0.f;
    #pragma unroll
    for (int t = 0; t < 4; ++t) {
      int d = lane * 4 + t;
      x1[t] = hrow[d];
      x2[t] = hrow[d + 128];
      ss += x1[t] * x1[t] + x2[t] * x2[t];
    }
    ss = red_sum32(ss);
    float rstd = rsqrtf(ss * (1.0f / HD_) + EPS_);
    __bf16* dst = kb + ((size_t)hk * S_LEN + s) * HD_;
    #pragma unroll
    for (int t = 0; t < 4; ++t) {
      int d = lane * 4 + t;
      float a = x1[t] * rstd * (1.0f + knw[d]);
      float b = x2[t] * rstd * (1.0f + knw[d + 128]);
      float c = fc[d], sn = fs[d];
      dst[d]       = (__bf16)(a * c - b * sn);
      dst[d + 128] = (__bf16)(a * sn + b * c);
    }
  }

  {
    const int hv = wid;
    const float* hrow = row + (NH_ + NKV_) * HD_ + hv * HD_;
    #pragma unroll
    for (int t = 0; t < 8; ++t) {
      int d = lane * 8 + t;
      vt[((size_t)hv * HD_ + d) * S_LEN + s] = (__bf16)hrow[d];
    }
  }
}

// ---------------- flash attention (sliding window + tanh softcap) ----------
__global__ __launch_bounds__(128)
void attn_kernel(const __bf16* __restrict__ qb, const __bf16* __restrict__ kbuf,
                 const __bf16* __restrict__ vt, __bf16* __restrict__ ob) {
  __shared__ float lds[4][16 * 32];

  const int lane = threadIdx.x & 31;
  const int wid  = threadIdx.x >> 5;
  const int h    = blockIdx.x;
  const int qblk = blockIdx.y * 4 + wid;
  const int qr0  = qblk * 16;
  const int kvh  = h >> 1;               // GQA: rep = NH/NKV = 2
  const int rlo  = lane & 15;
  const bool hi  = lane >= 16;
  const int kA0  = hi ? 8 : 0;
  const int kA1  = hi ? 24 : 16;

  const __bf16* qhead = qb   + ((size_t)h   * S_LEN + qr0) * HD_;
  const __bf16* khead = kbuf +  (size_t)kvh * S_LEN * HD_;
  const __bf16* vhead = vt   +  (size_t)kvh * HD_ * S_LEN;

  v16bf qa[8];
  {
    const __bf16* qrow = qhead + (size_t)rlo * HD_;
    #pragma unroll
    for (int f = 0; f < 8; ++f)
      qa[f] = load_frag(qrow + f * 32 + kA0, qrow + f * 32 + kA1);
  }

  v8f o[16];
  #pragma unroll
  for (int c = 0; c < 16; ++c) o[c] = (v8f){};
  float mrun[8], lrun[8];
  #pragma unroll
  for (int r = 0; r < 8; ++r) { mrun[r] = -3.0e38f; lrun[r] = 0.f; }

  int kstart = qr0 - (WINDOW_ - 1);
  if (kstart < 0) kstart = 0;
  kstart &= ~31;
  const int kend = qr0 + 16;

  float* plds = lds[wid];

  for (int kb = kstart; kb < kend; kb += 32) {
    v8f c0 = (v8f){}, c1 = (v8f){};
    const __bf16* krow0 = khead + (size_t)(kb + rlo)      * HD_;
    const __bf16* krow1 = khead + (size_t)(kb + 16 + rlo) * HD_;
    #pragma unroll
    for (int f = 0; f < 8; ++f) {
      v16bf b0 = load_frag(krow0 + f * 32 + kA0, krow0 + f * 32 + kA1);
      v16bf b1 = load_frag(krow1 + f * 32 + kA0, krow1 + f * 32 + kA1);
      c0 = __builtin_amdgcn_wmma_f32_16x16x32_bf16(false, qa[f], false, b0,
                                                   (short)0, c0, false, false);
      c1 = __builtin_amdgcn_wmma_f32_16x16x32_bf16(false, qa[f], false, b1,
                                                   (short)0, c1, false, false);
    }

    #pragma unroll
    for (int r = 0; r < 8; ++r) {
      const int row = hi ? r + 8 : r;
      const int i   = qr0 + row;
      const int j0  = kb + rlo;
      const int j1  = j0 + 16;
      float s0 = SOFTCAP_ * tanhf(c0[r] * (1.0f / SOFTCAP_));
      float s1 = SOFTCAP_ * tanhf(c1[r] * (1.0f / SOFTCAP_));
      s0 = ((j0 <= i) && (i - j0 < WINDOW_)) ? s0 : MASKV;
      s1 = ((j1 <= i) && (i - j1 < WINDOW_)) ? s1 : MASKV;
      float mx = red_max16(fmaxf(s0, s1));
      float mnew = fmaxf(mrun[r], mx);
      float alpha = __expf(mrun[r] - mnew);
      float p0 = __expf(s0 - mnew);
      float p1 = __expf(s1 - mnew);
      float rsum = red_sum16(p0 + p1);
      lrun[r] = lrun[r] * alpha + rsum;
      mrun[r] = mnew;
      #pragma unroll
      for (int ch = 0; ch < 16; ++ch) o[ch][r] *= alpha;
      plds[row * 32 + rlo]      = p0;
      plds[row * 32 + 16 + rlo] = p1;
    }

    v16bf pa;
    {
      const float* prow = plds + rlo * 32;
      #pragma unroll
      for (int i = 0; i < 8; ++i) {
        pa[i]     = (__bf16)prow[kA0 + i];
        pa[8 + i] = (__bf16)prow[kA1 + i];
      }
    }

    #pragma unroll
    for (int ch = 0; ch < 16; ++ch) {
      const __bf16* vrow = vhead + (size_t)(ch * 16 + rlo) * S_LEN + kb;
      v16bf vb = load_frag(vrow + kA0, vrow + kA1);
      o[ch] = __builtin_amdgcn_wmma_f32_16x16x32_bf16(false, pa, false, vb,
                                                      (short)0, o[ch], false, false);
    }
  }

  #pragma unroll
  for (int r = 0; r < 8; ++r) {
    const int row = hi ? r + 8 : r;
    const float inv = 1.0f / lrun[r];
    __bf16* orow = ob + (size_t)(qr0 + row) * O_IN + h * HD_;
    #pragma unroll
    for (int ch = 0; ch < 16; ++ch)
      orow[ch * 16 + rlo] = (__bf16)(o[ch][r] * inv);
  }
}

// ---------------- host launcher ----------------
extern "C" void kernel_launch(void* const* d_in, const int* in_sizes, int n_in,
                              void* d_out, int out_size, void* d_ws, size_t ws_size,
                              hipStream_t stream) {
  const float* x     = (const float*)d_in[0];
  const float* qkv_w = (const float*)d_in[1];
  const float* o_w   = (const float*)d_in[2];
  const float* qnw   = (const float*)d_in[3];
  const float* knw   = (const float*)d_in[4];
  const float* fcos  = (const float*)d_in[5];
  const float* fsin  = (const float*)d_in[6];
  float* out = (float*)d_out;

  size_t off = 0;
  auto alloc = [&](size_t bytes) -> void* {
    void* p = (char*)d_ws + off;
    off += (bytes + 255) & ~(size_t)255;
    return p;
  };
  __bf16* x_bf    = (__bf16*)alloc((size_t)S_LEN * HIDDEN_ * 2);
  __bf16* w_bf    = (__bf16*)alloc((size_t)QKV_OUT * HIDDEN_ * 2);
  __bf16* ow_bf   = (__bf16*)alloc((size_t)HIDDEN_ * O_IN * 2);
  float*  qkv     = (float*) alloc((size_t)S_LEN * QKV_OUT * 4);
  __bf16* q_bf    = (__bf16*)alloc((size_t)NH_  * S_LEN * HD_ * 2);
  __bf16* k_bf    = (__bf16*)alloc((size_t)NKV_ * S_LEN * HD_ * 2);
  __bf16* vt_bf   = (__bf16*)alloc((size_t)NKV_ * HD_ * S_LEN * 2);
  __bf16* attn_bf = (__bf16*)alloc((size_t)S_LEN * O_IN * 2);
  (void)ws_size; (void)n_in; (void)in_sizes; (void)out_size;

  cvt_bf16_kernel<<<2048, 256, 0, stream>>>(x, x_bf, S_LEN * HIDDEN_);
  cvt_bf16_kernel<<<4096, 256, 0, stream>>>(qkv_w, w_bf, QKV_OUT * HIDDEN_);
  cvt_bf16_kernel<<<2048, 256, 0, stream>>>(o_w, ow_bf, HIDDEN_ * O_IN);

  gemm_bf16_kernel<<<dim3(QKV_OUT / TN, S_LEN / TM), 256, 0, stream>>>(
      x_bf, w_bf, qkv, S_LEN, QKV_OUT, HIDDEN_);

  prep_kernel<<<S_LEN, 256, 0, stream>>>(qkv, qnw, knw, fcos, fsin,
                                         q_bf, k_bf, vt_bf);

  attn_kernel<<<dim3(NH_, S_LEN / 16 / 4), 128, 0, stream>>>(
      q_bf, k_bf, vt_bf, attn_bf);

  gemm_bf16_kernel<<<dim3(HIDDEN_ / TN, S_LEN / TM), 256, 0, stream>>>(
      attn_bf, ow_bf, out, S_LEN, HIDDEN_, O_IN);
}
